// MultiHeadAttention_37486474559682
// MI455X (gfx1250) — compile-verified, matrix operand order checked by reading, not run on hardware
//
#include <hip/hip_runtime.h>
#include <hip/hip_bf16.h>

typedef __attribute__((ext_vector_type(16))) __bf16 v16bf;
typedef __attribute__((ext_vector_type(8)))  __bf16 v8bf;
typedef __attribute__((ext_vector_type(4)))  __bf16 v4bf;
typedef __attribute__((ext_vector_type(8)))  float  v8f;
typedef __attribute__((ext_vector_type(4)))  float  v4f;
typedef __attribute__((ext_vector_type(4)))  int    v4i;

#define B_  2
#define S_  2048
#define H_  1024
#define NH_ 16
#define DK_ 64

// ---------------------------------------------------------------------------
// WMMA operand builder: CDNA5 16-bit A/B layout, lane-half h holds
// K{8h..8h+7} in elements 0..7 and K{16+8h..16+8h+7} in elements 8..15.
static __device__ __forceinline__ v16bf load_op16(const __bf16* base, int h) {
  v8bf lo = *(const v8bf*)(base + 8 * h);
  v8bf hi = *(const v8bf*)(base + 16 + 8 * h);
  v16bf r;
#pragma unroll
  for (int i = 0; i < 8; ++i) { r[i] = lo[i]; r[8 + i] = hi[i]; }
  return r;
}

// 16-byte async global->LDS copy (ASYNCcnt-tracked).
static __device__ __forceinline__ void async_copy16(const __bf16* g, __bf16* l) {
#if __has_builtin(__builtin_amdgcn_global_load_async_to_lds_b128)
  __builtin_amdgcn_global_load_async_to_lds_b128(
      (__attribute__((address_space(1))) v4i*)(uintptr_t)g,
      (__attribute__((address_space(3))) v4i*)(unsigned)(uintptr_t)l, 0, 0);
#else
  asm volatile("global_load_async_to_lds_b128 %0, %1, off"
               :: "v"((unsigned)(uintptr_t)l),
                  "v"((unsigned long long)(uintptr_t)g)
               : "memory");
#endif
}

static __device__ __forceinline__ void wait_async6() {
  asm volatile("s_wait_asynccnt 0x6" ::: "memory");
}
static __device__ __forceinline__ void wait_async0() {
  asm volatile("s_wait_asynccnt 0x0" ::: "memory");
}

__global__ void cvt_f32_bf16(const float* __restrict__ in,
                             __bf16* __restrict__ out, int n4) {
  int i = blockIdx.x * blockDim.x + threadIdx.x;
  int stride = gridDim.x * blockDim.x;
  for (; i < n4; i += stride) {
    v4f x = ((const v4f*)in)[i];
    v4bf y;
#pragma unroll
    for (int j = 0; j < 4; ++j) y[j] = (__bf16)x[j];
    ((v4bf*)out)[i] = y;
  }
}

// ---------------------------------------------------------------------------
// C[M,N] = A[M,K](bf16) @ W[N,K]^T(bf16) + bias[N](f32)
// LDS-staged, double-buffered async pipeline.  WG = 256 thr (8 waves),
// tile 64(M) x 128(N) x 64(K) per stage; wave tile 32x32; 8 WMMAs/stage/wave.
// mode 1: bf16 out [B,NH,S,DK]; mode 2: bf16 out [B,NH,DK,S]; mode 3: f32 [M,N]
#define LDS_STRIDE 72  // 64 + 8 bf16 pad -> 36-bank row stride, conflict-free

__global__ void __launch_bounds__(256)
gemm_bf16_wmma(const __bf16* __restrict__ A, const __bf16* __restrict__ W,
               const float* __restrict__ bias, void* __restrict__ out,
               int M, int N, int K, int mode) {
  __shared__ __bf16 sA[2][64][LDS_STRIDE];
  __shared__ __bf16 sB[2][128][LDS_STRIDE];

  const int tid  = threadIdx.x;
  const int wid  = tid >> 5;
  const int lane = tid & 31;
  const int lm = lane & 15, h = lane >> 4;
  const int wm = wid & 1, wn = wid >> 1;
  const int mBase = blockIdx.x * 64;
  const int nBase = blockIdx.y * 128;

  // staging: 4 threads/row, each thread covers elem [c8,c8+8) and [c8+32,c8+40)
  const int ra = tid >> 2;         // 0..63
  const int c8 = (tid & 3) * 8;
  const __bf16* gA  = A + (size_t)(mBase + ra) * K + c8;
  const __bf16* gB0 = W + (size_t)(nBase + ra) * K + c8;
  const __bf16* gB1 = W + (size_t)(nBase + 64 + ra) * K + c8;

  v8f acc[2][2] = {};

  const int NK = K / 64;  // stages

  auto stage = [&](int buf, int k0) {
    async_copy16(gA  + k0,      &sA[buf][ra][c8]);
    async_copy16(gA  + k0 + 32, &sA[buf][ra][c8 + 32]);
    async_copy16(gB0 + k0,      &sB[buf][ra][c8]);
    async_copy16(gB0 + k0 + 32, &sB[buf][ra][c8 + 32]);
    async_copy16(gB1 + k0,      &sB[buf][64 + ra][c8]);
    async_copy16(gB1 + k0 + 32, &sB[buf][64 + ra][c8 + 32]);
  };

  stage(0, 0);  // prologue

  for (int kc = 0; kc < NK; ++kc) {
    const int cur = kc & 1;
    if (kc + 1 < NK) {
      stage(cur ^ 1, (kc + 1) * 64);
      wait_async6();   // previous buffer's 6 copies (this wave) complete
    } else {
      wait_async0();
    }
    __syncthreads();   // all waves' copies into `cur` visible

    const __bf16* ap = &sA[cur][wm * 32 + lm][0];
    const __bf16* bp = &sB[cur][wn * 32 + lm][0];
#pragma unroll
    for (int kk = 0; kk < 2; ++kk) {
      v16bf a0 = load_op16(ap + kk * 32, h);
      v16bf a1 = load_op16(ap + 16 * LDS_STRIDE + kk * 32, h);
      v16bf b0 = load_op16(bp + kk * 32, h);
      v16bf b1 = load_op16(bp + 16 * LDS_STRIDE + kk * 32, h);
      acc[0][0] = __builtin_amdgcn_wmma_f32_16x16x32_bf16(false, a0, false, b0, (short)0, acc[0][0], false, false);
      acc[0][1] = __builtin_amdgcn_wmma_f32_16x16x32_bf16(false, a0, false, b1, (short)0, acc[0][1], false, false);
      acc[1][0] = __builtin_amdgcn_wmma_f32_16x16x32_bf16(false, a1, false, b0, (short)0, acc[1][0], false, false);
      acc[1][1] = __builtin_amdgcn_wmma_f32_16x16x32_bf16(false, a1, false, b1, (short)0, acc[1][1], false, false);
    }

    __syncthreads();   // reads of `cur` done before it is overwritten
  }

#pragma unroll
  for (int ms = 0; ms < 2; ++ms)
#pragma unroll
    for (int ns = 0; ns < 2; ++ns) {
      const int n = nBase + wn * 32 + ns * 16 + lm;
      const float bn = bias[n];
#pragma unroll
      for (int j = 0; j < 8; ++j) {
        const int m = mBase + wm * 32 + ms * 16 + j + 8 * h;
        const float v = acc[ms][ns][j] + bn;
        if (mode == 3) {
          ((float*)out)[(size_t)m * N + n] = v;
        } else {
          const int bb = m >> 11, s = m & (S_ - 1);   // m = b*S + s
          const int hd = n >> 6,  d = n & (DK_ - 1);  // n = h*DK + d
          __bf16* o = (__bf16*)out;
          if (mode == 1)
            o[(((size_t)(bb * NH_ + hd)) * S_ + s) * DK_ + d] = (__bf16)v;
          else
            o[(((size_t)(bb * NH_ + hd)) * DK_ + d) * S_ + s] = (__bf16)v;
        }
      }
    }
}

// ---------------------------------------------------------------------------
// Flash attention, software-pipelined: one wave owns a 16-row Q tile, streams
// keys in chunks of 32.  Next chunk's K operands are prefetched while softmax
// VALU and the P LDS round-trip run.  Q,K: [B,NH,S,DK]; V: [B,NH,DK,S] bf16.
__global__ void __launch_bounds__(256)
flash_attn_bf16(const __bf16* __restrict__ Q, const __bf16* __restrict__ Kh,
                const __bf16* __restrict__ Vt, const int* __restrict__ mask,
                __bf16* __restrict__ ctx) {
  __shared__ __bf16 ldsP[8][16 * 32];
  const int wid  = threadIdx.x >> 5;
  const int lane = threadIdx.x & 31;
  const int lm = lane & 15, h = lane >> 4;

  const int gt = blockIdx.x * 8 + wid;
  const int qt = gt & (S_ / 16 - 1);
  const int bh = gt >> 7;
  const int hd = bh & (NH_ - 1);
  const int bb = bh >> 4;
  const int qbase = qt * 16;

  const __bf16* Qp = Q  + ((size_t)(bb * NH_ + hd)) * S_ * DK_;
  const __bf16* Kp = Kh + ((size_t)(bb * NH_ + hd)) * S_ * DK_;
  const __bf16* Vp = Vt + ((size_t)(bb * NH_ + hd)) * DK_ * S_;
  const int* mp = mask + (size_t)bb * S_ * S_;

  v16bf aq[2];
#pragma unroll
  for (int c = 0; c < 2; ++c)
    aq[c] = load_op16(Qp + (size_t)(qbase + lm) * DK_ + 32 * c, h);

  v8f oacc[4] = {};
  float mrow[8], lrow[8];
#pragma unroll
  for (int j = 0; j < 8; ++j) { mrow[j] = -3.0e38f; lrow[j] = 0.f; }

  v16bf bk[2][2][2];  // [buf][key-subtile][dk-chunk]
  v16bf bv[4];
  int   mk[2][8];

  auto loadKtile = [&](int kb, int buf) {
#pragma unroll
    for (int t = 0; t < 2; ++t)
#pragma unroll
      for (int c = 0; c < 2; ++c)
        bk[buf][t][c] =
            load_op16(Kp + (size_t)(kb + 16 * t + lm) * DK_ + 32 * c, h);
  };

  loadKtile(0, 0);
  int pb = 0;
  for (int kb = 0; kb < S_; kb += 32, pb ^= 1) {
    // ---- scores: S[16q x 32k], K-dim = DK in two chunks of 32
    v8f sc[2] = {};
#pragma unroll
    for (int t = 0; t < 2; ++t)
#pragma unroll
      for (int c = 0; c < 2; ++c)
        sc[t] = __builtin_amdgcn_wmma_f32_16x16x32_bf16(
            false, aq[c], false, bk[pb][t][c], (short)0, sc[t], false, false);

    // ---- issue loads early (in-order completion => partial waits later):
    // mask (needed soonest), then V (needed after softmax), then next-K.
#pragma unroll
    for (int t = 0; t < 2; ++t)
#pragma unroll
      for (int j = 0; j < 8; ++j)
        mk[t][j] = mp[(size_t)(qbase + j + 8 * h) * S_ + kb + 16 * t + lm];
#pragma unroll
    for (int nt = 0; nt < 4; ++nt)
      bv[nt] = load_op16(Vp + (size_t)(nt * 16 + lm) * S_ + kb, h);
    if (kb + 32 < S_) loadKtile(kb + 32, pb ^ 1);

    // ---- scale by 1/sqrt(DK), apply mask
#pragma unroll
    for (int t = 0; t < 2; ++t)
#pragma unroll
      for (int j = 0; j < 8; ++j) {
        const float v = sc[t][j] * 0.125f;
        sc[t][j] = (mk[t][j] != 0) ? v : -60000.f;
      }

    // ---- online softmax; C-layout rows live in 16-lane halves
    float alpha[8];
#pragma unroll
    for (int j = 0; j < 8; ++j) {
      float cm = fmaxf(sc[0][j], sc[1][j]);
#pragma unroll
      for (int d = 1; d < 16; d <<= 1) cm = fmaxf(cm, __shfl_xor(cm, d, 32));
      const float mnew = fmaxf(mrow[j], cm);
      alpha[j] = __expf(mrow[j] - mnew);
      const float p0 = __expf(sc[0][j] - mnew);
      const float p1 = __expf(sc[1][j] - mnew);
      sc[0][j] = p0; sc[1][j] = p1;
      float rs = p0 + p1;
#pragma unroll
      for (int d = 1; d < 16; d <<= 1) rs += __shfl_xor(rs, d, 32);
      lrow[j] = lrow[j] * alpha[j] + rs;
      mrow[j] = mnew;
    }
#pragma unroll
    for (int nt = 0; nt < 4; ++nt)
#pragma unroll
      for (int j = 0; j < 8; ++j) oacc[nt][j] *= alpha[j];

    // ---- P: C-layout regs -> LDS row-major -> reload as A operand
    __bf16* pl = &ldsP[wid][0];
#pragma unroll
    for (int t = 0; t < 2; ++t)
#pragma unroll
      for (int j = 0; j < 8; ++j)
        pl[(j + 8 * h) * 32 + 16 * t + lm] = (__bf16)sc[t][j];
    asm volatile("s_wait_dscnt 0x0" ::: "memory");
    v16bf pa = load_op16(pl + lm * 32, h);

    // ---- O += P @ V (4 N-tiles over DK)
#pragma unroll
    for (int nt = 0; nt < 4; ++nt)
      oacc[nt] = __builtin_amdgcn_wmma_f32_16x16x32_bf16(
          false, pa, false, bv[nt], (short)0, oacc[nt], false, false);
  }

  // ---- normalize, emit context [B,S,H] bf16
#pragma unroll
  for (int j = 0; j < 8; ++j) {
    const float inv = 1.f / lrow[j];
    const int qr = qbase + j + 8 * h;
#pragma unroll
    for (int nt = 0; nt < 4; ++nt)
      ctx[((size_t)(bb * S_ + qr)) * H_ + hd * DK_ + nt * 16 + lm] =
          (__bf16)(oacc[nt][j] * inv);
  }
}

// ---------------------------------------------------------------------------
extern "C" void kernel_launch(void* const* d_in, const int* in_sizes, int n_in,
                              void* d_out, int out_size, void* d_ws, size_t ws_size,
                              hipStream_t stream) {
  (void)in_sizes; (void)n_in; (void)out_size; (void)ws_size;
  const float* query = (const float*)d_in[0];
  const float* key   = (const float*)d_in[1];
  const int*   amask = (const int*)d_in[2];
  const float* Wq = (const float*)d_in[3]; const float* bq = (const float*)d_in[4];
  const float* Wk = (const float*)d_in[5]; const float* bk = (const float*)d_in[6];
  const float* Wv = (const float*)d_in[7]; const float* bv = (const float*)d_in[8];
  const float* Wo = (const float*)d_in[9]; const float* bo = (const float*)d_in[10];

  char* ws = (char*)d_ws;
  const size_t MB = 1u << 20;
  const int SZ_X = B_ * S_ * H_;  // 4194304 elements
  const int SZ_W = H_ * H_;       // 1048576 elements
  __bf16* Xq  = (__bf16*)(ws +  0 * MB);
  __bf16* Xk  = (__bf16*)(ws +  8 * MB);
  __bf16* Wqb = (__bf16*)(ws + 16 * MB);
  __bf16* Wkb = (__bf16*)(ws + 18 * MB);
  __bf16* Wvb = (__bf16*)(ws + 20 * MB);
  __bf16* Wob = (__bf16*)(ws + 22 * MB);
  __bf16* Qh  = (__bf16*)(ws + 24 * MB);  // [B,NH,S,DK]
  __bf16* Khd = (__bf16*)(ws + 32 * MB);  // [B,NH,S,DK]
  __bf16* Vtd = (__bf16*)(ws + 40 * MB);  // [B,NH,DK,S]
  __bf16* ctx = (__bf16*)(ws + 48 * MB);  // [B,S,H]

  cvt_f32_bf16<<<1024, 256, 0, stream>>>(query, Xq, SZ_X / 4);
  cvt_f32_bf16<<<1024, 256, 0, stream>>>(key,   Xk, SZ_X / 4);
  cvt_f32_bf16<<<256,  256, 0, stream>>>(Wq, Wqb, SZ_W / 4);
  cvt_f32_bf16<<<256,  256, 0, stream>>>(Wk, Wkb, SZ_W / 4);
  cvt_f32_bf16<<<256,  256, 0, stream>>>(Wv, Wvb, SZ_W / 4);
  cvt_f32_bf16<<<256,  256, 0, stream>>>(Wo, Wob, SZ_W / 4);

  dim3 gg(64, 8);  // M/64 x N/128
  gemm_bf16_wmma<<<gg, 256, 0, stream>>>(Xq, Wqb, bq, Qh,  B_*S_, H_, H_, 1);
  gemm_bf16_wmma<<<gg, 256, 0, stream>>>(Xk, Wkb, bk, Khd, B_*S_, H_, H_, 1);
  gemm_bf16_wmma<<<gg, 256, 0, stream>>>(Xk, Wvb, bv, Vtd, B_*S_, H_, H_, 2);

  flash_attn_bf16<<<512, 256, 0, stream>>>(Qh, Khd, Vtd, amask, ctx);

  gemm_bf16_wmma<<<gg, 256, 0, stream>>>(ctx, Wob, bo, d_out, B_*S_, H_, H_, 3);
}
